// LinearAttention_88519275971067
// MI455X (gfx1250) — matrix operand order repaired and checked by reading, not compiled, not run on hardware
//
#include <hip/hip_runtime.h>
#include <hip/hip_bf16.h>

// ---------------------------------------------------------------------------
// LinearAttention on MI455X (gfx1250, wave32, WMMA)
//   B=4, N=4096, DIM=1024, H=16, HD=64
// Projection GEMMs (137 of 142 GFLOP) run on v_wmma_f32_16x16x32_bf16.
// Weights pre-converted to bf16 once; bf16 tiles staged into LDS with
// GLOBAL_LOAD_ASYNC_TO_LDS_B128 (ASYNCcnt path, no VGPR round-trip);
// f32 activation tiles convert in-flight. kv-einsum and ksum (k^T x ones)
// on WMMA; tiny q@kv apply in f32 VALU.
// ---------------------------------------------------------------------------

#define B_      4
#define N_      4096
#define DIM_    1024
#define H_      16
#define HD_     64
#define M_TOTAL (B_ * N_)      // 16384 rows through every projection

typedef __attribute__((ext_vector_type(16))) __bf16 v16bf;
typedef __attribute__((ext_vector_type(8)))  float  v8f;

__device__ __forceinline__ __bf16 f2bf(float f) { return (__bf16)f; }

// ---------------------------------------------------------------------------
// Async global -> LDS copy (CDNA5 §15.18.3 op 98, tracked by ASYNCcnt).
// VDST VGPR holds the per-lane LDS byte address; generic pointers to LDS
// carry the LDS offset in their low 32 bits (aperture in the high bits).
// ---------------------------------------------------------------------------
__device__ __forceinline__ void async_load_b128(void* lds_dst, const void* gsrc) {
    unsigned int lds = (unsigned int)(unsigned long long)lds_dst;
    asm volatile("global_load_async_to_lds_b128 %0, %1, off"
                 :: "v"(lds), "v"(gsrc)
                 : "memory");
}
__device__ __forceinline__ void wait_asynccnt0() {
    asm volatile("s_wait_asynccnt 0" ::: "memory");
}

// ---------------------------------------------------------------------------
// Fragment loaders per CDNA5 ISA 7.12.2 (wave32):
//  A 16x32 bf16: lane holds row m=lane%16; VGPR quad 0..3 -> K = q8..q8+7,
//                quad 4..7 -> K = 16+q8..16+q8+7, where q8 = (lane/16)*8.
//  B 32x16 bf16: lane holds col n=lane%16; same K order as A (q8.., 16+q8..).
//  C/D f32 16x16: element r -> m = r + 8*(lane/16), n = lane%16.
// Tiles are stored K-minor in LDS so these are contiguous/short reads.
// ---------------------------------------------------------------------------
__device__ __forceinline__ v16bf load_frag_a(const __bf16* p0, int ld, int lane) {
    const int m  = lane & 15;
    const int q8 = (lane >> 4) * 8;
    const __bf16* p = p0 + m * ld;
    v16bf r;
#pragma unroll
    for (int h = 0; h < 2; ++h) {
        const int kb = h * 16 + q8;
#pragma unroll
        for (int i = 0; i < 8; ++i) r[h * 8 + i] = p[kb + i];
    }
    return r;
}

__device__ __forceinline__ v16bf load_frag_b(const __bf16* p0, int ld, int lane) {
    const int n  = lane & 15;
    const int kq = (lane >> 4) * 8;
    const __bf16* p = p0 + n * ld + kq;
    v16bf r;
#pragma unroll
    for (int i = 0; i < 16; ++i) r[i] = p[(i >> 3) * 16 + (i & 7)];
    return r;
}

// ---------------------------------------------------------------------------
// One-time f32 -> bf16 weight conversion (each W element was otherwise
// re-converted by every M-block of the GEMM).
// ---------------------------------------------------------------------------
__global__ __launch_bounds__(256) void cvt_f32_bf16(
    const float* __restrict__ s, __bf16* __restrict__ d, int n)
{
    const int i = (blockIdx.x * 256 + threadIdx.x) * 8;
    if (i + 8 > n) return;
    float4 a = *(const float4*)(s + i);
    float4 b = *(const float4*)(s + i + 4);
    __bf16* dp = d + i;
    dp[0] = f2bf(a.x); dp[1] = f2bf(a.y); dp[2] = f2bf(a.z); dp[3] = f2bf(a.w);
    dp[4] = f2bf(b.x); dp[5] = f2bf(b.y); dp[6] = f2bf(b.z); dp[7] = f2bf(b.w);
}

// ---------------------------------------------------------------------------
// GEMM: Out[m,f] = act( X[m,:] . W[f,:] + bias[f] )   (Out = X @ W^T + b)
//   M=16384, N=1024, K=1024. Block tile 256x128, 8 waves (4Mx2N),
//   wave tile 64x64 (4x4 WMMA frags -> 16 WMMA per 8 fragment loads),
//   K step 32. Weight tile (bf16) staged via async global->LDS; activation
//   tile async too when already bf16, else f32 load + convert + ds_store.
//   ACT: 0 none, 1 relu, 2 relu + key_padding_mask (zero row's features).
//   OUT_BF16: store bf16 to workspace, else f32 (final output).
// ---------------------------------------------------------------------------
#define BM  256
#define BN  128
#define BK  32
#define LDT (BK + 8)   // 40 elems = 80B rows: 16B-aligned, bank-staggered

template <int IN_F32, int ACT, int OUT_BF16>
__global__ __launch_bounds__(256) void gemm_bias_act(
    const void* __restrict__ Xv, const __bf16* __restrict__ W,
    const float* __restrict__ bias, const int* __restrict__ mask,
    void* __restrict__ Outv)
{
    __shared__ __bf16 As[BM][LDT];   // activations tile [m][k]
    __shared__ __bf16 Bs[BN][LDT];   // weight tile      [f][k] (K-minor)

    const int t    = threadIdx.x;
    const int lane = t & 31;
    const int wave = t >> 5;
    const int wm   = wave & 3;    // 4 waves along M
    const int wn   = wave >> 2;   // 2 waves along N
    const int m0   = blockIdx.x * BM;
    const int n0   = blockIdx.y * BN;

    v8f acc[4][4];
#pragma unroll
    for (int i = 0; i < 4; ++i)
#pragma unroll
        for (int j = 0; j < 4; ++j)
#pragma unroll
            for (int e = 0; e < 8; ++e) acc[i][j][e] = 0.0f;

    const float*  Xf = (const float*)Xv;
    const __bf16* Xb = (const __bf16*)Xv;

    const int rowA = t;                 // A: one 32-elem row per thread
    const int rowB = t >> 1;            // B: 128 rows, 16 elems per thread
    const int koffB = (t & 1) * 16;

    for (int k0 = 0; k0 < DIM_; k0 += BK) {
        // ---- stage A tile ----
        if (IN_F32) {
            const float* src = Xf + (size_t)(m0 + rowA) * DIM_ + k0;
#pragma unroll
            for (int i = 0; i < 32; ++i) As[rowA][i] = f2bf(src[i]);
            if (k0 + BK < DIM_) __builtin_prefetch(src + BK, 0, 1);
        } else {
            const __bf16* src = Xb + (size_t)(m0 + rowA) * DIM_ + k0;
            async_load_b128(&As[rowA][0],  src);
            async_load_b128(&As[rowA][8],  src + 8);
            async_load_b128(&As[rowA][16], src + 16);
            async_load_b128(&As[rowA][24], src + 24);
        }
        // ---- stage B tile (weights, bf16): async global -> LDS ----
        {
            const __bf16* src = W + (size_t)(n0 + rowB) * DIM_ + k0 + koffB;
            async_load_b128(&Bs[rowB][koffB],     src);
            async_load_b128(&Bs[rowB][koffB + 8], src + 8);
            if (k0 + BK < DIM_) __builtin_prefetch(src + BK, 0, 1);
        }
        wait_asynccnt0();        // per-wave: async LDS writes landed
        __syncthreads();         // cross-wave visibility

        v16bf afr[4], bfr[4];
#pragma unroll
        for (int fm = 0; fm < 4; ++fm)
            afr[fm] = load_frag_a(&As[wm * 64 + fm * 16][0], LDT, lane);
#pragma unroll
        for (int fn = 0; fn < 4; ++fn)
            bfr[fn] = load_frag_b(&Bs[wn * 64 + fn * 16][0], LDT, lane);

#pragma unroll
        for (int fm = 0; fm < 4; ++fm)
#pragma unroll
            for (int fn = 0; fn < 4; ++fn)
                acc[fm][fn] = __builtin_amdgcn_wmma_f32_16x16x32_bf16(
                    false, afr[fm], false, bfr[fn], (short)0, acc[fm][fn],
                    false, false);
        __syncthreads();
    }

    // Epilogue: bias + activation (+mask) + store
    const int ln = lane & 15;
    const int lh = lane >> 4;
#pragma unroll
    for (int fm = 0; fm < 4; ++fm) {
        const int gmBase = m0 + wm * 64 + fm * 16;
#pragma unroll
        for (int fn = 0; fn < 4; ++fn) {
            const int gf = n0 + wn * 64 + fn * 16 + ln;
            const float bv = bias[gf];
#pragma unroll
            for (int r = 0; r < 8; ++r) {
                const int gm = gmBase + r + 8 * lh;   // flat (b*N + n)
                float vv = acc[fm][fn][r] + bv;
                if (ACT >= 1) vv = vv > 0.0f ? vv : 0.0f;
                if (ACT == 2) { if (mask[gm] != 0) vv = 0.0f; }
                if (OUT_BF16)
                    ((__bf16*)Outv)[(size_t)gm * DIM_ + gf] = f2bf(vv);
                else
                    ((float*)Outv)[(size_t)gm * DIM_ + gf] = vv;
            }
        }
    }
}

// ---------------------------------------------------------------------------
// kv[b,h,d,e] = sum_n k[b,n,h,d] * v[b,n,h,e];  ksum[b,h,d] = sum_n k[b,n,h,d]
// One block per (b,h); 8 waves each reduce a 512-row slab of N with WMMA
// (A = k^T tile, B = v tile, both staged K(=n)-minor in LDS; staging is a
// transpose/scatter so the async B128 path does not apply here).
// ksum is computed as k^T x ones with a constant all-ones B fragment,
// reusing the already-loaded A fragments. Cross-wave reduce via LDS atomics.
// ---------------------------------------------------------------------------
#define KVLD 34   // 32 + pad

__global__ __launch_bounds__(256) void kv_ksum_kernel(
    const __bf16* __restrict__ kbuf, const __bf16* __restrict__ vbuf,
    float* __restrict__ kvout, float* __restrict__ ksumout)
{
    __shared__ __bf16 kT[8][HD_][KVLD];   // [wave][d][n]  (transposed stage)
    __shared__ __bf16 vT[8][HD_][KVLD];   // [wave][e][n]
    __shared__ float  kvsh[HD_][HD_];
    __shared__ float  kssh[HD_];

    const int t    = threadIdx.x;
    const int lane = t & 31;
    const int wave = t >> 5;
    const int b    = blockIdx.x / H_;
    const int h    = blockIdx.x % H_;
    const size_t headOff = (size_t)h * HD_;

    for (int i = t; i < HD_ * HD_; i += 256) (&kvsh[0][0])[i] = 0.0f;
    if (t < HD_) kssh[t] = 0.0f;
    __syncthreads();

    v16bf bones;                 // all-ones B fragment for ksum = k^T x 1
#pragma unroll
    for (int i = 0; i < 16; ++i) bones[i] = (__bf16)1.0f;

    v8f acc[4][4], accks[4];
#pragma unroll
    for (int i = 0; i < 4; ++i) {
#pragma unroll
        for (int e = 0; e < 8; ++e) accks[i][e] = 0.0f;
#pragma unroll
        for (int j = 0; j < 4; ++j)
#pragma unroll
            for (int e = 0; e < 8; ++e) acc[i][j][e] = 0.0f;
    }

    const int nBeg = wave * (N_ / 8);
    for (int ns = 0; ns < N_ / 8; ns += 32) {
        const int n = nBeg + ns + lane;          // row this lane stages
        const __bf16* krow = kbuf + ((size_t)(b * N_ + n)) * DIM_ + headOff;
        const __bf16* vrow = vbuf + ((size_t)(b * N_ + n)) * DIM_ + headOff;
#pragma unroll
        for (int d = 0; d < HD_; ++d) {
            kT[wave][d][lane] = krow[d];
            vT[wave][d][lane] = vrow[d];
        }
        __syncthreads();

        v16bf afr[4], bfr[4];
#pragma unroll
        for (int dt = 0; dt < 4; ++dt)
            afr[dt] = load_frag_a(&kT[wave][dt * 16][0], KVLD, lane);
#pragma unroll
        for (int et = 0; et < 4; ++et)
            bfr[et] = load_frag_b(&vT[wave][et * 16][0], KVLD, lane);

#pragma unroll
        for (int dt = 0; dt < 4; ++dt) {
#pragma unroll
            for (int et = 0; et < 4; ++et)
                acc[dt][et] = __builtin_amdgcn_wmma_f32_16x16x32_bf16(
                    false, afr[dt], false, bfr[et], (short)0, acc[dt][et],
                    false, false);
            accks[dt] = __builtin_amdgcn_wmma_f32_16x16x32_bf16(
                false, afr[dt], false, bones, (short)0, accks[dt],
                false, false);
        }
        __syncthreads();
    }

    // cross-wave reduction in LDS
    const int ln = lane & 15;
    const int lh = lane >> 4;
#pragma unroll
    for (int dt = 0; dt < 4; ++dt)
#pragma unroll
        for (int et = 0; et < 4; ++et)
#pragma unroll
            for (int r = 0; r < 8; ++r)
                atomicAdd(&kvsh[dt * 16 + r + 8 * lh][et * 16 + ln],
                          acc[dt][et][r]);
    if (ln == 0) {               // every column of accks holds the same sums
#pragma unroll
        for (int dt = 0; dt < 4; ++dt)
#pragma unroll
            for (int r = 0; r < 8; ++r)
                atomicAdd(&kssh[dt * 16 + r + 8 * lh], accks[dt][r]);
    }
    __syncthreads();

    float* kvg = kvout + ((size_t)(b * H_ + h)) * HD_ * HD_;
    for (int i = t; i < HD_ * HD_; i += 256) kvg[i] = (&kvsh[0][0])[i];
    if (t < HD_) ksumout[(b * H_ + h) * HD_ + t] = kssh[t];
}

// ---------------------------------------------------------------------------
// out[b,n,h,e] = (sum_d q[b,n,h,d]*kv[b,h,d,e]) / (q . ksum + 1e-6)
// One block per (b,h, 256-row slab); kv/ksum cached in LDS; f32 VALU
// (only ~2 GFLOP; keeping f32 here preserves precision for free).
// ---------------------------------------------------------------------------
__global__ __launch_bounds__(256) void attn_apply_kernel(
    const __bf16* __restrict__ qbuf, const float* __restrict__ kv,
    const float* __restrict__ ksum, __bf16* __restrict__ att)
{
    __shared__ float kvs[HD_][HD_];
    __shared__ float kss[HD_];

    const int t = threadIdx.x;
    const int b = blockIdx.x / H_;
    const int h = blockIdx.x % H_;

    const float* kvg = kv + ((size_t)(b * H_ + h)) * HD_ * HD_;
    for (int i = t; i < HD_ * HD_; i += 256) (&kvs[0][0])[i] = kvg[i];
    if (t < HD_) kss[t] = ksum[(b * H_ + h) * HD_ + t];
    __syncthreads();

    const int n = blockIdx.y * 256 + t;
    const __bf16* qrow = qbuf + ((size_t)(b * N_ + n)) * DIM_ + h * HD_;
    float qf[HD_];
#pragma unroll
    for (int d = 0; d < HD_; ++d) qf[d] = (float)qrow[d];

    float dot = 0.0f;
#pragma unroll
    for (int d = 0; d < HD_; ++d) dot += qf[d] * kss[d];
    const float denom = 1.0f / (dot + 1e-6f);

    __bf16* orow = att + ((size_t)(b * N_ + n)) * DIM_ + h * HD_;
#pragma unroll 4
    for (int e = 0; e < HD_; ++e) {
        float a = 0.0f;
#pragma unroll
        for (int d = 0; d < HD_; ++d) a += qf[d] * kvs[d][e];  // lane-uniform addr -> LDS broadcast
        orow[e] = f2bf(a * denom);
    }
}

// ---------------------------------------------------------------------------
// Launch pipeline
// ---------------------------------------------------------------------------
extern "C" void kernel_launch(void* const* d_in, const int* in_sizes, int n_in,
                              void* d_out, int out_size, void* d_ws, size_t ws_size,
                              hipStream_t stream)
{
    (void)in_sizes; (void)n_in; (void)out_size; (void)ws_size;

    const float* query = (const float*)d_in[0];
    const float* key_  = (const float*)d_in[1];
    const float* value = (const float*)d_in[2];
    const float* Wq    = (const float*)d_in[3];
    const float* bq    = (const float*)d_in[4];
    const float* Wk    = (const float*)d_in[5];
    const float* bk    = (const float*)d_in[6];
    const float* Wv    = (const float*)d_in[7];
    const float* bv    = (const float*)d_in[8];
    const float* Wo    = (const float*)d_in[9];
    const float* bo    = (const float*)d_in[10];
    const int*   mask  = (const int*)d_in[11];   // key_padding_mask [B,N]
    // d_in[12] = need_weights (ignored; reference returns out only)

    char*  ws = (char*)d_ws;
    const size_t actBytes = (size_t)M_TOTAL * DIM_ * sizeof(__bf16);  // 32 MB
    const size_t wBytes   = (size_t)DIM_ * DIM_ * sizeof(__bf16);     // 2 MB
    __bf16* qb   = (__bf16*)(ws);
    __bf16* kb   = (__bf16*)(ws + actBytes);
    __bf16* vb   = (__bf16*)(ws + 2 * actBytes);
    __bf16* ab   = (__bf16*)(ws + 3 * actBytes);
    char*   p    = ws + 4 * actBytes;
    __bf16* Wqb  = (__bf16*)(p);             p += wBytes;
    __bf16* Wkb  = (__bf16*)(p);             p += wBytes;
    __bf16* Wvb  = (__bf16*)(p);             p += wBytes;
    __bf16* Wob  = (__bf16*)(p);             p += wBytes;
    float*  kv   = (float*)(p);              p += (size_t)B_ * H_ * HD_ * HD_ * sizeof(float);
    float*  ksum = (float*)(p);

    dim3 blk(256);

    // one-time weight conversion (f32 -> bf16)
    const int wElems = DIM_ * DIM_;
    dim3 gCvt(wElems / (256 * 8));
    cvt_f32_bf16<<<gCvt, blk, 0, stream>>>(Wq, Wqb, wElems);
    cvt_f32_bf16<<<gCvt, blk, 0, stream>>>(Wk, Wkb, wElems);
    cvt_f32_bf16<<<gCvt, blk, 0, stream>>>(Wv, Wvb, wElems);
    cvt_f32_bf16<<<gCvt, blk, 0, stream>>>(Wo, Wob, wElems);

    dim3 gGemm(M_TOTAL / BM, DIM_ / BN);   // 64 x 8 blocks

    gemm_bias_act<1, 1, 1><<<gGemm, blk, 0, stream>>>(query, Wqb, bq, nullptr, qb);
    gemm_bias_act<1, 2, 1><<<gGemm, blk, 0, stream>>>(key_,  Wkb, bk, mask,    kb);
    gemm_bias_act<1, 0, 1><<<gGemm, blk, 0, stream>>>(value, Wvb, bv, nullptr, vb);

    kv_ksum_kernel<<<dim3(B_ * H_), blk, 0, stream>>>(kb, vb, kv, ksum);

    attn_apply_kernel<<<dim3(B_ * H_, N_ / 256), blk, 0, stream>>>(qb, kv, ksum, ab);

    gemm_bias_act<0, 0, 0><<<gGemm, blk, 0, stream>>>(ab, Wob, bo, nullptr, d_out);
}
